// GraphWalkerMemory_64828236366326
// MI455X (gfx1250) — matrix operand order repaired and checked by reading, not compiled, not run on hardware
//
#include <hip/hip_runtime.h>
#include <hip/hip_bf16.h>
#include <math.h>

// Problem constants (match reference)
#define NWALK 131072
#define NPLANES 8
#define DIN 192
#define DHID 256
#define DOUT 128
#define TILE_M 64   // rows per block: 4x B-fragment register reuse

typedef __bf16 v16bf __attribute__((ext_vector_type(16)));
typedef __bf16 v8bf  __attribute__((ext_vector_type(8)));
typedef float  v8f   __attribute__((ext_vector_type(8)));

// ---------------------------------------------------------------------------
// Pass 0: zero the per-plane counters (must run every launch: deterministic)
// ---------------------------------------------------------------------------
__global__ void ppmlp_zero_counts(int* __restrict__ counts) {
    if (threadIdx.x < NPLANES) counts[threadIdx.x] = 0;
}

// ---------------------------------------------------------------------------
// Pass 1: bucket walker row indices by plane (atomic append)
// ---------------------------------------------------------------------------
__global__ void ppmlp_partition(const int* __restrict__ plane_idx,
                                int* __restrict__ counts,
                                int* __restrict__ idxlist) {
    int i = blockIdx.x * blockDim.x + threadIdx.x;
    if (i < NWALK) {
        int p = plane_idx[i] & (NPLANES - 1);
        int pos = atomicAdd(&counts[p], 1);
        idxlist[p * NWALK + pos] = i;
    }
}

// ---------------------------------------------------------------------------
// Pass 2: convert fp32 weights -> bf16 (weights are tiny: ~2.6MB fp32)
// ---------------------------------------------------------------------------
__global__ void ppmlp_convert_weights(const float* __restrict__ W1,
                                      const float* __restrict__ W2,
                                      __bf16* __restrict__ Wb1,
                                      __bf16* __restrict__ Wb2) {
    const int n1 = NPLANES * DHID * DIN;
    const int n2 = NPLANES * DOUT * DHID;
    const int stride = gridDim.x * blockDim.x;
    int i = blockIdx.x * blockDim.x + threadIdx.x;
    for (int e = i; e < n1; e += stride) Wb1[e] = (__bf16)W1[e];
    for (int e = i; e < n2; e += stride) Wb2[e] = (__bf16)W2[e];
}

// ---------------------------------------------------------------------------
// Pass 3: gathered per-plane 2-layer MLP on 64-row tiles via bf16 WMMA.
// Block = 256 threads = 8 waves; block owns 64 rows (4 M-tiles of 16).
//   Layer1: H[64x256] = gelu(X[64x192] * W1^T + b1)
//           wave w owns cols 32w..32w+31 (2 N-tiles); B-frags reused over 4 M
//   Layer2: O[64x128] = H * W2^T + b2
//           wave w owns cols 16w..16w+15 (1 N-tile);  B-frags reused over 4 M
// Weight traffic per block: W1[p]+W2[p] bf16 = 160KB amortized over 64 rows.
// ---------------------------------------------------------------------------
__global__ __launch_bounds__(256) void ppmlp_mlp(
    const float* __restrict__ x,
    const float* __restrict__ b1,
    const float* __restrict__ b2,
    const __bf16* __restrict__ Wb1,
    const __bf16* __restrict__ Wb2,
    const int* __restrict__ counts,
    const int* __restrict__ idxlist,
    float* __restrict__ out) {

    const int p    = blockIdx.y;
    const int cnt  = counts[p];
    const int tile = blockIdx.x;
    if (tile * TILE_M >= cnt) return;  // uniform per block -> EXEC all-1 for WMMA

    __shared__ __bf16 Xs[TILE_M][200]; // 192 cols, padded (400B rows, 16B-aligned)
    __shared__ __bf16 Hs[TILE_M][264]; // 256 cols, padded (528B rows, 16B-aligned)
    __shared__ int rowid[TILE_M];
    __shared__ int rvalid[TILE_M];

    const int tid = threadIdx.x;

    if (tid < TILE_M) {
        int g = tile * TILE_M + tid;
        int ok = (g < cnt) ? 1 : 0;
        rowid[tid]  = ok ? idxlist[p * NWALK + g] : 0;
        rvalid[tid] = ok;
    }
    __syncthreads();

    // Gather 64 rows of x into LDS as bf16 (256 threads x 48 elements)
    for (int e = tid; e < TILE_M * DIN; e += 256) {
        int r = e / DIN, c = e % DIN;
        Xs[r][c] = (__bf16)x[(size_t)rowid[r] * DIN + c];
    }
    __syncthreads();

    const int wave = tid >> 5;
    const int lane = tid & 31;
    const int ln   = lane & 15;   // M (A-frag) / N (B,C frags)
    const int hi   = lane >> 4;   // lane half selects K/M sub-range
    const int koff = hi * 8;      // unused; A/B frags take K from k0 + hi * 16

    union AFrag { v16bf v; struct { v8bf lo, hi; } s; };

    // ---------------- Layer 1: acc[4 M-tiles][2 N-tiles] --------------------
    v8f acc[4][2];
    #pragma unroll
    for (int t = 0; t < 2; ++t) {
        float bv = b1[p * DHID + wave * 32 + t * 16 + ln];
        #pragma unroll
        for (int mi = 0; mi < 4; ++mi)
            #pragma unroll
            for (int r = 0; r < 8; ++r) acc[mi][t][r] = bv;
    }

    for (int kk = 0; kk < DIN / 32; ++kk) {
        const int k0 = kk * 32;
        // A fragments for the 4 M-tiles (from LDS)
        AFrag a[4];
        #pragma unroll
        for (int mi = 0; mi < 4; ++mi) {
            a[mi].s.lo = *(const v8bf*)&Xs[mi * 16 + ln][k0 + hi * 16];
            a[mi].s.hi = *(const v8bf*)&Xs[mi * 16 + ln][k0 + hi * 16 + 8];
        }
        // B fragments for this wave's 2 N-tiles (from global, L2-resident)
        v16bf b[2];
        #pragma unroll
        for (int t = 0; t < 2; ++t) {
            const int n = wave * 32 + t * 16 + ln;
            b[t] = *(const v16bf*)(Wb1 + ((size_t)p * DHID + n) * DIN + k0 + hi * 16);
        }
        #pragma unroll
        for (int mi = 0; mi < 4; ++mi)
            #pragma unroll
            for (int t = 0; t < 2; ++t)
                acc[mi][t] = __builtin_amdgcn_wmma_f32_16x16x32_bf16(
                    false, a[mi].v, false, b[t], (short)0, acc[mi][t], false, false);
    }

    // Exact GELU (matches approximate=False) + write H tiles to LDS as bf16.
    // C/D layout: VGPR r, lane -> (M = mi*16 + r + 8*hi, N = ln)
    #pragma unroll
    for (int t = 0; t < 2; ++t) {
        const int n = wave * 32 + t * 16 + ln;
        #pragma unroll
        for (int mi = 0; mi < 4; ++mi)
            #pragma unroll
            for (int r = 0; r < 8; ++r) {
                float v = acc[mi][t][r];
                float g = 0.5f * v * (1.0f + erff(v * 0.70710678118654752f));
                Hs[mi * 16 + r + hi * 8][n] = (__bf16)g;
            }
    }
    __syncthreads();

    // ---------------- Layer 2: acc2[4 M-tiles], one N-tile per wave ---------
    const int n2 = wave * 16 + ln;
    v8f acc2[4];
    {
        float bv = b2[p * DOUT + n2];
        #pragma unroll
        for (int mi = 0; mi < 4; ++mi)
            #pragma unroll
            for (int r = 0; r < 8; ++r) acc2[mi][r] = bv;
    }
    for (int kk = 0; kk < DHID / 32; ++kk) {
        const int k0 = kk * 32;
        AFrag a[4];
        #pragma unroll
        for (int mi = 0; mi < 4; ++mi) {
            a[mi].s.lo = *(const v8bf*)&Hs[mi * 16 + ln][k0 + hi * 16];
            a[mi].s.hi = *(const v8bf*)&Hs[mi * 16 + ln][k0 + hi * 16 + 8];
        }
        v16bf b = *(const v16bf*)(Wb2 + ((size_t)p * DOUT + n2) * DHID + k0 + hi * 16);
        #pragma unroll
        for (int mi = 0; mi < 4; ++mi)
            acc2[mi] = __builtin_amdgcn_wmma_f32_16x16x32_bf16(
                false, a[mi].v, false, b, (short)0, acc2[mi], false, false);
    }

    // Masked scatter of f32 results back to out[row][n2]
    #pragma unroll
    for (int mi = 0; mi < 4; ++mi)
        #pragma unroll
        for (int r = 0; r < 8; ++r) {
            const int m = mi * 16 + r + hi * 8;
            if (rvalid[m]) out[(size_t)rowid[m] * DOUT + n2] = acc2[mi][r];
        }
}

// ---------------------------------------------------------------------------
// Launch: zero counters -> partition -> weight convert -> tiled WMMA MLP.
// Workspace layout:
//   [0,   64)           : int counts[8]
//   [64,  64+4MiB)      : int idxlist[8][131072]
//   [..,  +768KiB)      : bf16 Wb1[8][256][192]
//   [..,  +512KiB)      : bf16 Wb2[8][128][256]
// ---------------------------------------------------------------------------
extern "C" void kernel_launch(void* const* d_in, const int* in_sizes, int n_in,
                              void* d_out, int out_size, void* d_ws, size_t ws_size,
                              hipStream_t stream) {
    const float* x         = (const float*)d_in[0];
    const int*   plane_idx = (const int*)d_in[1];
    const float* W1        = (const float*)d_in[2];
    const float* b1        = (const float*)d_in[3];
    const float* W2        = (const float*)d_in[4];
    const float* b2        = (const float*)d_in[5];
    float*       out       = (float*)d_out;

    int*    counts  = (int*)d_ws;
    int*    idxlist = (int*)((char*)d_ws + 64);
    __bf16* Wb1     = (__bf16*)((char*)d_ws + 64 + (size_t)NPLANES * NWALK * sizeof(int));
    __bf16* Wb2     = Wb1 + (size_t)NPLANES * DHID * DIN;

    ppmlp_zero_counts<<<1, 32, 0, stream>>>(counts);
    ppmlp_partition<<<(NWALK + 255) / 256, 256, 0, stream>>>(plane_idx, counts, idxlist);
    ppmlp_convert_weights<<<512, 256, 0, stream>>>(W1, W2, Wb1, Wb2);
    ppmlp_mlp<<<dim3(NWALK / TILE_M, NPLANES), 256, 0, stream>>>(
        x, b1, b2, Wb1, Wb2, counts, idxlist, out);
}
